// LSAStage_82248623718729
// MI455X (gfx1250) — matrix operand order repaired and checked by reading, not compiled, not run on hardware
//
#include <hip/hip_runtime.h>
#include <hip/hip_bf16.h>

// ---------------------------------------------------------------------------
// Swin window-attention block (2 layers) for MI455X / gfx1250.
// All GEMM-shaped math runs through v_wmma_f32_16x16x32_bf16 (wave32 WMMA,
// f32 accumulate). f32 activations/weights are converted to bf16 for A/B.
// Weights are pre-transposed once to [N][K] so both A and B tiles stage into
// LDS with coalesced b128 loads (no per-block transpose work).
// ---------------------------------------------------------------------------

typedef __attribute__((ext_vector_type(16))) __bf16 v16bf;
typedef __attribute__((ext_vector_type(8)))  float  v8f;

__device__ __forceinline__ v8f wmma_bf16(v16bf a, v16bf b, v8f c) {
  // (neg_a, A, neg_b, B, c_mod, C, reuse_a, reuse_b)
  return __builtin_amdgcn_wmma_f32_16x16x32_bf16(false, a, false, b, (short)0, c,
                                                 false, false);
}

__device__ __forceinline__ v8f zero_v8f() {
  v8f z = {0.f, 0.f, 0.f, 0.f, 0.f, 0.f, 0.f, 0.f};
  return z;
}

union FragU { uint4 u[2]; v16bf v; };

// A operand (16x32 bf16, MxK): lanes 0-15 hold M=0..15, lanes 16-31 same M,
// other K-half. Per ISA layout each lane needs two 16B chunks at bf16 offsets
// m*ld + 8*half and m*ld + 16 + 8*half (ld must be a multiple of 8 bf16).
__device__ __forceinline__ v16bf load_frag_a(const __bf16* base, int lane, int ld) {
  const int half = lane >> 4, m = lane & 15;
  const __bf16* p = base + m * ld + 8 * half;
  FragU f;
  f.u[0] = *(const uint4*)(p);
  f.u[1] = *(const uint4*)(p + 16);
  return f.v;
}

// B operand (32x16 bf16, KxN) from LDS stored transposed as [n][k]:
// lane needs K = 8*half..8*half+7 and 16+8*half..16+8*half+7 for column
// n = lane&15 (same K order as A), i.e. two b128 reads.
__device__ __forceinline__ v16bf load_frag_b(const __bf16* base, int lane, int ld) {
  const int half = lane >> 4, n = lane & 15;
  const __bf16* p = base + n * ld + 8 * half;
  FragU f;
  f.u[0] = *(const uint4*)(p);
  f.u[1] = *(const uint4*)(p + 16);
  return f.v;
}

// ---------------------------------------------------------------------------
// Weight conversion f32 [L][K][N] -> bf16 transposed [L][N][K].
// One-time cost; makes every GEMM B-tile stage a coalesced b128 copy.
// ---------------------------------------------------------------------------
__global__ __launch_bounds__(256) void cvt_transpose_kernel(
    const float* __restrict__ in, unsigned short* __restrict__ out, int K, int N,
    int total) {
  const int id = blockIdx.x * 256 + threadIdx.x;
  if (id >= total) return;
  const int n = id % N;
  const int k = (id / N) % K;
  const int l = id / (N * K);
  out[((size_t)l * N + n) * K + k] = __builtin_bit_cast(unsigned short, (__bf16)in[id]);
}

// ---------------------------------------------------------------------------
// Window partition: x(B,56,56,384) -> xw(1024 windows, 49 tokens, 384) f32
// ---------------------------------------------------------------------------
__global__ __launch_bounds__(256) void partition_kernel(const float* __restrict__ x,
                                                        float* __restrict__ xw) {
  const int id = blockIdx.x * 256 + threadIdx.x;
  const int total = 50176 * 96;
  if (id >= total) return;
  const int c4 = (id % 96) * 4;
  const int row = id / 96;
  const int token = row % 49, win = row / 49;
  const int wb = win & 7, hb = (win >> 3) & 7, b = win >> 6;
  const int wx = token % 7, hy = token / 7;
  const int Hp = hb * 7 + hy, Wp = wb * 7 + wx;
  const float4 v = *(const float4*)(x + (((size_t)b * 56 + Hp) * 56 + Wp) * 384 + c4);
  *(float4*)(xw + (size_t)row * 384 + c4) = v;
}

__global__ __launch_bounds__(256) void merge_kernel(const float* __restrict__ xw,
                                                    float* __restrict__ out) {
  const int id = blockIdx.x * 256 + threadIdx.x;
  const int total = 50176 * 96;
  if (id >= total) return;
  const int c4 = (id % 96) * 4;
  const int row = id / 96;
  const int token = row % 49, win = row / 49;
  const int wb = win & 7, hb = (win >> 3) & 7, b = win >> 6;
  const int wx = token % 7, hy = token / 7;
  const int Hp = hb * 7 + hy, Wp = wb * 7 + wx;
  const float4 v = *(const float4*)(xw + (size_t)row * 384 + c4);
  *(float4*)(out + (((size_t)b * 56 + Hp) * 56 + Wp) * 384 + c4) = v;
}

// ---------------------------------------------------------------------------
// LayerNorm over C=384. One wave (32 lanes) per token; 12 elements per lane.
// ---------------------------------------------------------------------------
__global__ __launch_bounds__(256) void ln_kernel(const float* __restrict__ xw,
                                                 const float* __restrict__ sc,
                                                 const float* __restrict__ bi,
                                                 unsigned short* __restrict__ outb) {
  const int tid = threadIdx.x, wid = tid >> 5, lane = tid & 31;
  const size_t token = (size_t)blockIdx.x * 8 + wid;
  const float* row = xw + token * 384;
  float x[12];
  float sum = 0.f;
#pragma unroll
  for (int i = 0; i < 12; ++i) {
    x[i] = row[lane + 32 * i];
    sum += x[i];
  }
#pragma unroll
  for (int o = 16; o > 0; o >>= 1) sum += __shfl_xor(sum, o, 32);
  const float mu = sum * (1.0f / 384.0f);
  float vs = 0.f;
#pragma unroll
  for (int i = 0; i < 12; ++i) {
    const float d = x[i] - mu;
    vs += d * d;
  }
#pragma unroll
  for (int o = 16; o > 0; o >>= 1) vs += __shfl_xor(vs, o, 32);
  const float inv = rsqrtf(vs * (1.0f / 384.0f) + 1e-6f);
#pragma unroll
  for (int i = 0; i < 12; ++i) {
    const int c = lane + 32 * i;
    const float h = (x[i] - mu) * inv * sc[c] + bi[c];
    outb[token * 384 + c] = __builtin_bit_cast(unsigned short, (__bf16)h);
  }
}

// ---------------------------------------------------------------------------
// Tiled bf16 WMMA GEMM: C[M,N] = A[M,K] * Bt[N,K]^T (+bias) (+epilogue)
// A is [M][K] bf16 row-major; Bt is the pre-transposed weight [N][K] bf16.
// BM=128, BN=128, BK=32. 8 waves/block, each wave owns a 32x64 tile
// (8 v_wmma per K-step). Double-buffered LDS, padded stride 40 bf16 for
// conflict-free b128 fragment reads.
// EPI: 0 = store bf16, 1 = bias+GELU -> bf16, 2 = (+bias) += into f32 resid
// ---------------------------------------------------------------------------
template <int EPI>
__global__ __launch_bounds__(256) void gemm_bf16_kernel(
    const unsigned short* __restrict__ Abits, const unsigned short* __restrict__ Btbits,
    const float* __restrict__ bias, float* __restrict__ resid,
    unsigned short* __restrict__ outb, int K, int N) {
  constexpr int BM = 128, BN = 128, BK = 32, LD = 40;
  __shared__ __bf16 As[2][BM * LD];
  __shared__ __bf16 Bs[2][BN * LD];
  const __bf16* A = (const __bf16*)Abits;
  const __bf16* Bt = (const __bf16*)Btbits;

  const int tid = threadIdx.x;
  const int m0 = blockIdx.x * BM;
  const int n0 = blockIdx.y * BN;
  const int wid = tid >> 5, lane = tid & 31;
  const int wm = wid & 3, wn = wid >> 2;

  // Both tiles are 128 rows x 32 k-values of bf16, [row][k] layout.
  auto stage = [&](int k0, int buf) {
    const int r = tid >> 1, hh = tid & 1;
    {
      const uint4* ga = (const uint4*)(A + (size_t)(m0 + r) * K + k0 + hh * 16);
      uint4 a0 = ga[0];
      uint4 a1 = ga[1];
      __bf16* as = As[buf] + r * LD + hh * 16;
      *(uint4*)(as) = a0;
      *(uint4*)(as + 8) = a1;
    }
    {
      const uint4* gb = (const uint4*)(Bt + (size_t)(n0 + r) * K + k0 + hh * 16);
      uint4 b0 = gb[0];
      uint4 b1 = gb[1];
      __bf16* bs = Bs[buf] + r * LD + hh * 16;
      *(uint4*)(bs) = b0;
      *(uint4*)(bs + 8) = b1;
    }
    if (k0 + 2 * BK < K) {  // prefetch two K-steps ahead (global_prefetch_b8)
      __builtin_prefetch(A + (size_t)(m0 + r) * K + k0 + 2 * BK, 0, 1);
      __builtin_prefetch(Bt + (size_t)(n0 + r) * K + k0 + 2 * BK, 0, 1);
    }
  };

  v8f acc[2][4];
#pragma unroll
  for (int i = 0; i < 2; ++i)
#pragma unroll
    for (int j = 0; j < 4; ++j) acc[i][j] = zero_v8f();

  const int nk = K / BK;
  stage(0, 0);
  for (int kt = 0; kt < nk; ++kt) {
    __syncthreads();
    if (kt + 1 < nk) stage((kt + 1) * BK, (kt + 1) & 1);
    const __bf16* as = As[kt & 1];
    const __bf16* bs = Bs[kt & 1];
    v16bf a0 = load_frag_a(as + (wm * 32) * LD, lane, LD);
    v16bf a1 = load_frag_a(as + (wm * 32 + 16) * LD, lane, LD);
#pragma unroll
    for (int j = 0; j < 4; ++j) {
      v16bf b = load_frag_b(bs + (wn * 64 + j * 16) * LD, lane, LD);
      acc[0][j] = wmma_bf16(a0, b, acc[0][j]);
      acc[1][j] = wmma_bf16(a1, b, acc[1][j]);
    }
  }

  // Epilogue. C/D layout: VGPR r -> row r (lanes 0-15) / r+8 (lanes 16-31),
  // col = lane & 15.
  const int half = lane >> 4, nn = lane & 15;
#pragma unroll
  for (int i = 0; i < 2; ++i)
#pragma unroll
    for (int j = 0; j < 4; ++j)
#pragma unroll
      for (int r = 0; r < 8; ++r) {
        const int row = m0 + wm * 32 + i * 16 + r + 8 * half;
        const int col = n0 + wn * 64 + j * 16 + nn;
        float v = acc[i][j][r];
        if constexpr (EPI == 0) {
          outb[(size_t)row * N + col] = __builtin_bit_cast(unsigned short, (__bf16)v);
        } else if constexpr (EPI == 1) {
          v += bias[col];
          const float g =
              0.5f * v * (1.0f + tanhf(0.7978845608028654f * (v + 0.044715f * v * v * v)));
          outb[(size_t)row * N + col] = __builtin_bit_cast(unsigned short, (__bf16)g);
        } else {
          if (bias) v += bias[col];
          resid[(size_t)row * N + col] += v;
        }
      }
}

// ---------------------------------------------------------------------------
// Fused window attention: one block per (window, head). N=49 padded to 64,
// hd=32. S = q k^T (WMMA), scale + rel-pos bias + softmax in LDS, O = P v
// (WMMA, K=64 -> two 32-steps). Output bf16 for the proj GEMM.
// ---------------------------------------------------------------------------
__global__ __launch_bounds__(256) void attn_kernel(const unsigned short* __restrict__ qkvbits,
                                                   const float* __restrict__ rpb,
                                                   unsigned short* __restrict__ obits) {
  constexpr int LDQ = 40, LDV = 72, LDSF = 68, LDP = 72;
  __shared__ __bf16 Qs[64 * LDQ];
  __shared__ __bf16 Ks[64 * LDQ];
  __shared__ __bf16 Vt[32 * LDV];  // v transposed: [dim][token]
  __shared__ float Sf[64 * LDSF];
  __shared__ __bf16 Pb[64 * LDP];

  const __bf16* qkv = (const __bf16*)qkvbits;
  const int tid = threadIdx.x;
  const int w = blockIdx.x, h = blockIdx.y;

  // Stage q, k, v^T (zero-pad tokens 49..63).
  for (int i = tid; i < 64 * 32; i += 256) {
    const int t = i >> 5, d = i & 31;
    __bf16 qv = (__bf16)0.f, kv = (__bf16)0.f, vv = (__bf16)0.f;
    if (t < 49) {
      const size_t base = ((size_t)w * 49 + t) * 1152 + h * 32 + d;
      qv = qkv[base];
      kv = qkv[base + 384];
      vv = qkv[base + 768];
    }
    Qs[t * LDQ + d] = qv;
    Ks[t * LDQ + d] = kv;
    Vt[d * LDV + t] = vv;
  }
  __syncthreads();

  const int wid = tid >> 5, lane = tid & 31;
  const int half = lane >> 4, nn = lane & 15;

  // S = q @ k^T : 4x4 tiles of 16x16, 2 tiles per wave, single K=32 step.
  {
    const int sm = wid & 3, sn = wid >> 2;
    v16bf aq = load_frag_a(Qs + sm * 16 * LDQ, lane, LDQ);
#pragma unroll
    for (int j = 0; j < 2; ++j) {
      const int nt = sn * 2 + j;
      v16bf bk = load_frag_b(Ks + nt * 16 * LDQ, lane, LDQ);
      v8f s = zero_v8f();
      s = wmma_bf16(aq, bk, s);
#pragma unroll
      for (int r = 0; r < 8; ++r)
        Sf[(sm * 16 + r + 8 * half) * LDSF + nt * 16 + nn] = s[r];
    }
  }
  __syncthreads();

  // Softmax rows (scale + relative position bias), one row per thread.
  if (tid < 64) {
    const int row = tid;
    if (row < 49) {
      const int qy = row / 7, qx = row % 7;
      const float scale = 0.17677669529663687f;  // 32^-0.5
      float mx = -3.0e38f;
      for (int c = 0; c < 49; ++c) {
        const int ky = c / 7, kx = c % 7;
        const int ridx = (qy - ky + 6) * 13 + (qx - kx + 6);
        const float v = Sf[row * LDSF + c] * scale + rpb[ridx * 12 + h];
        Sf[row * LDSF + c] = v;
        mx = fmaxf(mx, v);
      }
      float sum = 0.f;
      for (int c = 0; c < 49; ++c) {
        const float e = expf(Sf[row * LDSF + c] - mx);
        Sf[row * LDSF + c] = e;
        sum += e;
      }
      const float inv = 1.0f / sum;
      for (int c = 0; c < 64; ++c) {
        const float p = (c < 49) ? Sf[row * LDSF + c] * inv : 0.f;
        Pb[row * LDP + c] = (__bf16)p;
      }
    } else {
      for (int c = 0; c < 64; ++c) Pb[row * LDP + c] = (__bf16)0.f;
    }
  }
  __syncthreads();

  // O = P @ v : 4x2 tiles of 16x16, one per wave, K=64 -> two 32-steps.
  {
    const int om = wid & 3, on = (wid >> 2) & 1;
    v8f acc = zero_v8f();
#pragma unroll
    for (int ks = 0; ks < 2; ++ks) {
      v16bf ap = load_frag_a(Pb + om * 16 * LDP + ks * 32, lane, LDP);
      v16bf bv = load_frag_b(Vt + on * 16 * LDV + ks * 32, lane, LDV);
      acc = wmma_bf16(ap, bv, acc);
    }
#pragma unroll
    for (int r = 0; r < 8; ++r) {
      const int t = om * 16 + r + 8 * half;
      if (t < 49)
        obits[((size_t)w * 49 + t) * 384 + h * 32 + on * 16 + nn] =
            __builtin_bit_cast(unsigned short, (__bf16)acc[r]);
    }
  }
}

// ---------------------------------------------------------------------------
// Host orchestration
// ---------------------------------------------------------------------------
extern "C" void kernel_launch(void* const* d_in, const int* in_sizes, int n_in,
                              void* d_out, int out_size, void* d_ws, size_t ws_size,
                              hipStream_t stream) {
  const float* x = (const float*)d_in[0];
  const float* ln1_s = (const float*)d_in[1];
  const float* ln1_b = (const float*)d_in[2];
  const float* qkv_w = (const float*)d_in[3];
  const float* rpb = (const float*)d_in[4];
  const float* proj_w = (const float*)d_in[5];
  const float* ln2_s = (const float*)d_in[6];
  const float* ln2_b = (const float*)d_in[7];
  const float* mlp_w1 = (const float*)d_in[8];
  const float* mlp_b1 = (const float*)d_in[9];
  const float* mlp_w2 = (const float*)d_in[10];
  const float* mlp_b2 = (const float*)d_in[11];

  char* ws = (char*)d_ws;
  const size_t M = 50176;  // 1024 windows * 49 tokens
  size_t off = 0;
  float* xw = (float*)(ws + off);                off += M * 384 * 4;
  unsigned short* hbf = (unsigned short*)(ws + off);  off += M * 384 * 2;
  unsigned short* obf = (unsigned short*)(ws + off);  off += M * 384 * 2;
  // qkv (M x 1152) and MLP hidden (M x 1536) have disjoint live ranges: alias.
  unsigned short* big = (unsigned short*)(ws + off);  off += M * 1536 * 2;
  unsigned short* qkvw = (unsigned short*)(ws + off); off += (size_t)2 * 384 * 1152 * 2;
  unsigned short* projw = (unsigned short*)(ws + off); off += (size_t)2 * 384 * 384 * 2;
  unsigned short* w1w = (unsigned short*)(ws + off);  off += (size_t)2 * 384 * 1536 * 2;
  unsigned short* w2w = (unsigned short*)(ws + off);  off += (size_t)2 * 1536 * 384 * 2;

  // Convert + transpose weights: [L][K][N] f32 -> [L][N][K] bf16.
  cvt_transpose_kernel<<<(2 * 384 * 1152) / 256, 256, 0, stream>>>(
      qkv_w, qkvw, 384, 1152, 2 * 384 * 1152);
  cvt_transpose_kernel<<<(2 * 384 * 384) / 256, 256, 0, stream>>>(
      proj_w, projw, 384, 384, 2 * 384 * 384);
  cvt_transpose_kernel<<<(2 * 384 * 1536) / 256, 256, 0, stream>>>(
      mlp_w1, w1w, 384, 1536, 2 * 384 * 1536);
  cvt_transpose_kernel<<<(2 * 1536 * 384) / 256, 256, 0, stream>>>(
      mlp_w2, w2w, 1536, 384, 2 * 1536 * 384);

  partition_kernel<<<(50176 * 96) / 256, 256, 0, stream>>>(x, xw);

  for (int l = 0; l < 2; ++l) {
    ln_kernel<<<6272, 256, 0, stream>>>(xw, ln1_s + l * 384, ln1_b + l * 384, hbf);
    gemm_bf16_kernel<0><<<dim3(392, 9), 256, 0, stream>>>(
        hbf, qkvw + (size_t)l * 384 * 1152, nullptr, nullptr, big, 384, 1152);
    attn_kernel<<<dim3(1024, 12), 256, 0, stream>>>(big, rpb + l * 169 * 12, obf);
    gemm_bf16_kernel<2><<<dim3(392, 3), 256, 0, stream>>>(
        obf, projw + (size_t)l * 384 * 384, nullptr, xw, nullptr, 384, 384);
    ln_kernel<<<6272, 256, 0, stream>>>(xw, ln2_s + l * 384, ln2_b + l * 384, hbf);
    gemm_bf16_kernel<1><<<dim3(392, 12), 256, 0, stream>>>(
        hbf, w1w + (size_t)l * 384 * 1536, mlp_b1 + l * 1536, nullptr, big, 384, 1536);
    gemm_bf16_kernel<2><<<dim3(392, 3), 256, 0, stream>>>(
        big, w2w + (size_t)l * 1536 * 384, mlp_b2 + l * 384, xw, nullptr, 1536, 384);
  }

  merge_kernel<<<(50176 * 96) / 256, 256, 0, stream>>>(xw, (float*)d_out);

  (void)in_sizes; (void)n_in; (void)out_size; (void)ws_size;
}
